// TaskNearestAssigner_79164837200686
// MI455X (gfx1250) — matrix operand order repaired and checked by reading, not compiled, not run on hardware
//
#include <hip/hip_runtime.h>
#include <hip/hip_bf16.h>

// ---------------------------------------------------------------------------
// TaskNearestAssigner for MI455X (gfx1250, wave32).
// Bandwidth-bound: 561MB of output, floor ~25us @ 23.3 TB/s.
// scores (516MB) are synthesized as one-hot tiles on the WMMA pipe.
// ---------------------------------------------------------------------------

#define BSZ   16
#define MGT   256
#define NCLS  80
#define NANC  3
#define NG    33600            // 160^2 + 80^2 + 40^2

// flat float offsets into d_out (tuple concatenated in return order)
#define OFF_TXY  ((size_t)0)
#define OFF_WH   ((size_t)3225600)          // 16*3*33600*2
#define OFF_SC   ((size_t)6451200)
#define OFF_ANC  ((size_t)135475200)        // OFF_SC + 16*3*33600*80
#define OFF_MASK ((size_t)138700800)

typedef __attribute__((ext_vector_type(16))) _Float16 v16h;
typedef __attribute__((ext_vector_type(8)))  float    v8f;

// ---------------------------------------------------------------------------
// Exact top-3 nearest cells by L1 metric |i+0.5-gx|+|j+0.5-gy|.
// Any cell with |di|+|dj|>=3 from the containing cell has metric >= 2, which
// bounds the in-window 3rd-best, so a 5x5 window suffices. Ties broken by
// lower linear index g = j*S+i, matching lax.top_k.
// ---------------------------------------------------------------------------
__device__ __forceinline__ void top3_cells(float gx, float gy, int S, int g_out[3]) {
    int ix = (int)floorf(gx);
    int iy = (int)floorf(gy);
    float m0 = 1e30f, m1 = 1e30f, m2 = 1e30f;
    int   g0 = 0x40000000, g1 = 0x40000000, g2 = 0x40000000;
    for (int dj = -2; dj <= 2; ++dj) {
        int j = iy + dj;
        if (j < 0 || j >= S) continue;
        for (int di = -2; di <= 2; ++di) {
            int i = ix + di;
            if (i < 0 || i >= S) continue;
            float met = fabsf((float)i + 0.5f - gx) + fabsf((float)j + 0.5f - gy);
            int   g   = j * S + i;
            if (met < m0 || (met == m0 && g < g0)) {
                m2 = m1; g2 = g1; m1 = m0; g1 = g0; m0 = met; g0 = g;
            } else if (met < m1 || (met == m1 && g < g1)) {
                m2 = m1; g2 = g1; m1 = met; g1 = g;
            } else if (met < m2 || (met == m2 && g < g2)) {
                m2 = met; g2 = g;
            }
        }
    }
    g_out[0] = g0; g_out[1] = g1; g_out[2] = g2;
}

// ---------------------------------------------------------------------------
// K0: init per-cell claim state
// ---------------------------------------------------------------------------
__global__ void k_init(unsigned* __restrict__ count, unsigned* __restrict__ minm) {
    int i = blockIdx.x * blockDim.x + threadIdx.x;
    if (i < BSZ * NG) { count[i] = 0u; minm[i] = 0xFFFFFFFFu; }
}

// ---------------------------------------------------------------------------
// K1: per-GT analytic top-3 candidates, scattered with atomics.
// count == fg (pre-conflict claim count); minm == first claimant (== argmax
// of a 0/1 mask over m, which picks the lowest set index).
// ---------------------------------------------------------------------------
__global__ void k_candidates(const float* __restrict__ gt_cxy,
                             const float* __restrict__ mask_gt,
                             unsigned* __restrict__ count,
                             unsigned* __restrict__ minm) {
    int t = blockIdx.x * blockDim.x + threadIdx.x;
    if (t >= BSZ * MGT) return;
    int b = t / MGT, m = t % MGT;
    if (!(mask_gt[t] > 0.0f)) return;
    float cx = gt_cxy[2 * t], cy = gt_cxy[2 * t + 1];
    const int   Ss[3]   = {160, 80, 40};
    const float inv[3]  = {0.125f, 0.0625f, 0.03125f};
    const int   loff[3] = {0, 25600, 32000};
    for (int l = 0; l < 3; ++l) {
        int g3[3];
        top3_cells(cx * inv[l], cy * inv[l], Ss[l], g3);
        for (int k = 0; k < 3; ++k) {
            int idx = b * NG + loff[l] + g3[k];
            atomicAdd(&count[idx], 1u);
            atomicMin(&minm[idx], (unsigned)m);
        }
    }
}

// ---------------------------------------------------------------------------
// K2: per-cell conflict resolution + compact target record.
// Level boundaries (25600, 32000) are multiples of 256, so each block is
// single-level; GT centers (scaled by 1/stride) staged in LDS.
// ---------------------------------------------------------------------------
__global__ void k_resolve(const float* __restrict__ gt_cxy,
                          const float* __restrict__ gt_wh,
                          const int*   __restrict__ gt_labels,
                          const float* __restrict__ mask_gt,
                          const unsigned* __restrict__ count,
                          const unsigned* __restrict__ minm,
                          float4* __restrict__ rec4,
                          int2*   __restrict__ reci) {
    int b  = blockIdx.y;
    int G0 = blockIdx.x * blockDim.x;
    int G  = G0 + threadIdx.x;

    int l = (G0 < 25600) ? 0 : (G0 < 32000) ? 1 : 2;
    int   S    = (l == 0) ? 160 : (l == 1) ? 80 : 40;
    float invs = (l == 0) ? 0.125f : (l == 1) ? 0.0625f : 0.03125f;
    int   loff = (l == 0) ? 0 : (l == 1) ? 25600 : 32000;

    __shared__ float2 sc[MGT];
    {
        float2 c = ((const float2*)gt_cxy)[b * MGT + threadIdx.x];
        sc[threadIdx.x] = make_float2(c.x * invs, c.y * invs);
    }
    __syncthreads();
    if (G >= NG) return;

    int local = G - loff;
    int x = local % S, y = local / S;
    int idx = b * NG + G;

    unsigned cnt = count[idx];
    int tgt = 0, fg = 0;
    if (cnt == 1u) {
        tgt = (int)minm[idx];
        fg  = 1;
    } else if (cnt > 1u) {
        // faithful quirk: argmax of metric over ALL gts (first max on ties)
        float cxc = (float)x + 0.5f, cyc = (float)y + 0.5f;
        float best = -1.0f; int mam = 0;
        for (int m = 0; m < MGT; ++m) {
            float d = fabsf(cxc - sc[m].x) + fabsf(cyc - sc[m].y);
            if (d > best) { best = d; mam = m; }
        }
        int g3[3];
        top3_cells(sc[mam].x, sc[mam].y, S, g3);
        bool member = (g3[0] == local) | (g3[1] == local) | (g3[2] == local);
        if (member && (mask_gt[b * MGT + mam] > 0.0f)) { tgt = mam; fg = 1; }
    }

    float2 c = sc[tgt];
    float2 w = ((const float2*)gt_wh)[b * MGT + tgt];
    rec4[idx] = make_float4(c.x - (float)x, c.y - (float)y, w.x * invs, w.y * invs);
    reci[idx] = make_int2(gt_labels[b * MGT + tgt], fg);
}

// ---------------------------------------------------------------------------
// K3: one-hot scores via WMMA. One wave per (batch, 16-cell tile).
// scores[g, 16j+n] = delta(label_g, 16j+n) = (A x B)[g,n] with
//   A : 16x32 f16 one-hot of (label-16j) with k = s + 16*(lane/16), same order as B
//   B : 32x16 f16 identity slice (lanes 0-15, slot s == lane)
// C layout: VGPR r <-> row r (+8 for upper half-wave), lane <-> column, so
// each per-VGPR store is two contiguous 64B row segments. Wave always fully
// active (EXEC all ones, required for WMMA).
// ---------------------------------------------------------------------------
__global__ void k_scores(const int2* __restrict__ reci, float* __restrict__ out_sc) {
    int wave = (blockIdx.x * blockDim.x + threadIdx.x) >> 5;
    int lane = threadIdx.x & 31;
    int b    = wave / (NG / 16);
    int tile = wave % (NG / 16);
    int g0   = tile * 16;
    int n    = lane & 15;                 // column within 16-col tile; also row id for A data
    bool hi  = lane >= 16;

    int lab = reci[b * NG + g0 + n].x;    // label of row (lane&15)

    v16h Bm;
#pragma unroll
    for (int s = 0; s < 16; ++s)
        Bm[s] = (!hi && s == lane) ? (_Float16)1.0f : (_Float16)0.0f;

#pragma unroll
    for (int j = 0; j < 5; ++j) {
        int sel = lab - 16 * j;
        v16h Am;
#pragma unroll
        for (int s = 0; s < 16; ++s) {
            // k order used here: lanes 0-15: K = s
            //                  lanes16-31: K = s+16
            int K = hi ? s + 16 : s;
            Am[s] = (K == sel) ? (_Float16)1.0f : (_Float16)0.0f;
        }
        v8f C = {};
        C = __builtin_amdgcn_wmma_f32_16x16x32_f16(false, Am, false, Bm,
                                                   (short)0, C, false, false);
#pragma unroll
        for (int r = 0; r < 8; ++r) {
            int row = r + (hi ? 8 : 0);
            int g   = g0 + row;
            float v = C[r];
#pragma unroll
            for (int a = 0; a < 3; ++a) {
                size_t o = ((size_t)((b * 3 + a) * NG + g)) * NCLS + 16 * j + n;
                out_sc[o] = v;
            }
        }
    }
}

// ---------------------------------------------------------------------------
// K4: txy / wh / anch / mask fill from compact records (broadcast over na=3).
// ---------------------------------------------------------------------------
__global__ void k_small(const float4* __restrict__ rec4,
                        const int2*   __restrict__ reci,
                        const float*  __restrict__ anc_whs,
                        float* __restrict__ out) {
    int t = blockIdx.x * blockDim.x + threadIdx.x;
    if (t >= BSZ * NANC * NG) return;
    int g  = t % NG;
    int ba = t / NG;
    int a  = ba % NANC;
    int b  = ba / NANC;
    int l = (g < 25600) ? 0 : (g < 32000) ? 1 : 2;
    float invs = (l == 0) ? 0.125f : (l == 1) ? 0.0625f : 0.03125f;

    float4 r  = rec4[b * NG + g];
    int2   ri = reci[b * NG + g];
    float  ax = anc_whs[(l * 3 + a) * 2 + 0] * invs;
    float  ay = anc_whs[(l * 3 + a) * 2 + 1] * invs;

    float rx = r.z / ax, ry = r.w / ay;
    float mx = fmaxf(rx, 1.0f / rx), my = fmaxf(ry, 1.0f / ry);
    float mk = (ri.y > 0 && fmaxf(mx, my) < 4.0f) ? 1.0f : 0.0f;

    ((float2*)(out + OFF_TXY))[t] = make_float2(r.x, r.y);
    ((float2*)(out + OFF_WH ))[t] = make_float2(r.z, r.w);
    ((float2*)(out + OFF_ANC))[t] = make_float2(ax, ay);
    out[OFF_MASK + t] = mk;
}

// ---------------------------------------------------------------------------
extern "C" void kernel_launch(void* const* d_in, const int* in_sizes, int n_in,
                              void* d_out, int out_size, void* d_ws, size_t ws_size,
                              hipStream_t stream) {
    const float* anc_whs   = (const float*)d_in[0];
    // d_in[1..3] = grids (regenerated analytically)
    const int*   gt_labels = (const int*)  d_in[4];
    const float* gt_cxy    = (const float*)d_in[5];
    const float* gt_wh     = (const float*)d_in[6];
    // d_in[7] = strides (8/16/32, compiled in)
    const float* mask_gt   = (const float*)d_in[8];
    float* out = (float*)d_out;

    // workspace: count | minm | rec4 | reci  (~17.2 MB)
    unsigned* count = (unsigned*)d_ws;
    unsigned* minm  = count + (size_t)BSZ * NG;
    float4*   rec4  = (float4*)(minm + (size_t)BSZ * NG);   // 16B-aligned
    int2*     reci  = (int2*)(rec4 + (size_t)BSZ * NG);

    k_init      <<<(BSZ * NG + 255) / 256, 256, 0, stream>>>(count, minm);
    k_candidates<<<(BSZ * MGT + 255) / 256, 256, 0, stream>>>(gt_cxy, mask_gt, count, minm);
    k_resolve   <<<dim3(132, BSZ), 256, 0, stream>>>(gt_cxy, gt_wh, gt_labels, mask_gt,
                                                     count, minm, rec4, reci);
    k_scores    <<<(BSZ * (NG / 16)) / 8, 256, 0, stream>>>(reci, out + OFF_SC);
    k_small     <<<(BSZ * NANC * NG + 255) / 256, 256, 0, stream>>>(rec4, reci, anc_whs, out);
}
